// ActorNetwork_66314295050521
// MI455X (gfx1250) — compile-verified, matrix operand order checked by reading, not run on hardware
//
#include <hip/hip_runtime.h>
#include <hip/hip_bf16.h>

// ---------------------------------------------------------------------------
// Types for CDNA5 WMMA (wave32): 16x16x32 bf16 -> f32
// ---------------------------------------------------------------------------
typedef __bf16  bf16_t;
typedef bf16_t  v16bf __attribute__((ext_vector_type(16)));
typedef float   v8f   __attribute__((ext_vector_type(8)));

#define DGCN 128

// ---------------------------------------------------------------------------
// Packed-fragment index helpers (CDNA5 ISA 7.12.2, 16-bit A 16x32 layout):
//   lane L (half = L>>4, r = L&15):
//     v0..v3 hold K = 8*half + 0..7 ; v4..v7 hold K = 16 + 8*half + 0..7
// Packed buffer: [tile][kstep][lane][16 bf16] -> every fragment is one
// contiguous, 32B-aligned chunk per lane (coalesced 1KB per wave).
// ---------------------------------------------------------------------------
__device__ __forceinline__ void k_to_half_j(int kr, int& half, int& j) {
    if (kr < 8)       { half = 0; j = kr; }
    else if (kr < 16) { half = 1; j = kr - 8; }
    else if (kr < 24) { half = 0; j = 8 + (kr - 16); }
    else              { half = 1; j = 8 + (kr - 24); }
}

__device__ __forceinline__ size_t packA_idx(int m, int k, int K) {
    int tm = m >> 4, mr = m & 15;
    int s = k >> 5, kr = k & 31;
    int half, j; k_to_half_j(kr, half, j);
    int L = half * 16 + mr;
    return ((size_t)(tm * (K >> 5) + s) * 32 + L) * 16 + j;
}

__device__ __forceinline__ size_t packB_idx(int k, int n, int K) {
    int tn = n >> 4, nr = n & 15;
    int s = k >> 5, kr = k & 31;
    int half, j; k_to_half_j(kr, half, j);
    int L = half * 16 + nr;
    return ((size_t)(tn * (K >> 5) + s) * 32 + L) * 16 + j;
}

// ---------------------------------------------------------------------------
// Small elementwise / graph kernels
// ---------------------------------------------------------------------------
__global__ void set_ones_kernel(float* __restrict__ p, int n) {
    int i = blockIdx.x * blockDim.x + threadIdx.x;
    if (i < n) p[i] = 1.0f;
}

__global__ void deg_count_kernel(const long long* __restrict__ dst,
                                 float* __restrict__ deg, int E) {
    int i = blockIdx.x * blockDim.x + threadIdx.x;
    if (i < E) atomicAdd(&deg[(int)dst[i]], 1.0f);
}

__global__ void rsqrt_inplace_kernel(float* __restrict__ p, int n) {
    int i = blockIdx.x * blockDim.x + threadIdx.x;
    if (i < n) p[i] = rsqrtf(p[i]);   // deg >= 1 (self loop)
}

// f32 row-major -> packed-A bf16 fragments
__global__ void pack_a_f32_kernel(const float* __restrict__ in,
                                  bf16_t* __restrict__ out, int M, int K) {
    size_t i = (size_t)blockIdx.x * blockDim.x + threadIdx.x;
    if (i < (size_t)M * K) {
        int m = (int)(i / K);
        int k = (int)(i - (size_t)m * K);
        out[packA_idx(m, k, K)] = (bf16_t)in[i];
    }
}

// f32 row-major KxN weight -> packed-B bf16 fragments
__global__ void pack_b_f32_kernel(const float* __restrict__ in,
                                  bf16_t* __restrict__ out, int K, int N) {
    size_t i = (size_t)blockIdx.x * blockDim.x + threadIdx.x;
    if (i < (size_t)K * N) {
        int k = (int)(i / N);
        int n = (int)(i - (size_t)k * N);
        out[packB_idx(k, n, K)] = (bf16_t)in[i];
    }
}

// out[i,k] = h[i,k] * dinv[i]^2   (self-loop message seed)
__global__ void init_out_kernel(const float* __restrict__ h,
                                const float* __restrict__ dinv,
                                float* __restrict__ out, size_t total) {
    size_t i = (size_t)blockIdx.x * blockDim.x + threadIdx.x;
    if (i < total) {
        int node = (int)(i >> 7);
        float di = dinv[node];
        out[i] = h[i] * di * di;
    }
}

// One wave (32 lanes) per edge: float4 gather of h[src], scaled atomic
// scatter-add into out[dst]. out (51 MB) is L2-resident (192 MB).
__global__ void scatter_kernel(const long long* __restrict__ src,
                               const long long* __restrict__ dst,
                               const float* __restrict__ h,
                               const float* __restrict__ dinv,
                               float* __restrict__ out, int E) {
    int wid  = (int)(((size_t)blockIdx.x * blockDim.x + threadIdx.x) >> 5);
    int lane = threadIdx.x & 31;
    if (wid >= E) return;
    int s = (int)src[wid];
    int d = (int)dst[wid];
    float norm = dinv[s] * dinv[d];
    const float4* hrow = (const float4*)(h + (size_t)s * DGCN);
    float4 v = hrow[lane];
    float* orow = out + (size_t)d * DGCN + lane * 4;
    atomicAdd(orow + 0, v.x * norm);
    atomicAdd(orow + 1, v.y * norm);
    atomicAdd(orow + 2, v.z * norm);
    atomicAdd(orow + 3, v.w * norm);
}

// a_packedA = bf16( relu(out + b_gcn) + x ) in packed-A layout (K = 128)
__global__ void relu_residual_pack_kernel(const float* __restrict__ out,
                                          const float* __restrict__ bgcn,
                                          const float* __restrict__ x,
                                          bf16_t* __restrict__ apk, size_t total) {
    size_t i = (size_t)blockIdx.x * blockDim.x + threadIdx.x;
    if (i < total) {
        int m = (int)(i >> 7);
        int k = (int)(i & (DGCN - 1));
        float v = fmaxf(out[i] + bgcn[k], 0.0f) + x[i];
        apk[packA_idx(m, k, DGCN)] = (bf16_t)v;
    }
}

// d_out[i] = dot(h2[i, 0:256], W3) + b3 ; one wave per row (wave32 shuffles)
__global__ void final_dot_kernel(const bf16_t* __restrict__ h2,
                                 const float* __restrict__ W3,
                                 const float* __restrict__ b3,
                                 float* __restrict__ out, int N) {
    int wid  = (int)(((size_t)blockIdx.x * blockDim.x + threadIdx.x) >> 5);
    int lane = threadIdx.x & 31;
    if (wid >= N) return;
    const bf16_t* row = h2 + (size_t)wid * 256;
    float acc = 0.0f;
#pragma unroll
    for (int j = 0; j < 8; ++j) {
        int k = lane + 32 * j;
        acc += (float)row[k] * W3[k];
    }
#pragma unroll
    for (int off = 16; off > 0; off >>= 1)
        acc += __shfl_down(acc, off, 32);
    if (lane == 0) out[wid] = acc + b3[0];
}

// ---------------------------------------------------------------------------
// WMMA GEMM on packed fragments. One wave computes a 64x16 output strip
// (4 m-subtiles x 1 n-tile). Software-pipelined: k-step s+1's fragments are
// in flight while step s's four WMMAs execute, so the pre-WMMA wait only
// covers already-old loads. Packed-A is padded to the grid footprint ->
// branch-free k-loop; remainder handled by scalar-guarded stores only.
//   OUT_MODE 0: f32 row-major   1: bf16 packed-A (next GEMM, K_next = N)
//   OUT_MODE 2: bf16 row-major
// ---------------------------------------------------------------------------
template<int OUT_MODE, bool HAS_BIAS, bool RELU>
__device__ __forceinline__ void store_subtile(const v8f& c, int m0, int n, int half,
                                              const float* __restrict__ bias,
                                              float* __restrict__ Cf,
                                              bf16_t* __restrict__ Cb, int N) {
    float bv = HAS_BIAS ? bias[n] : 0.0f;
#pragma unroll
    for (int r = 0; r < 8; ++r) {
        int m = m0 + r + 8 * half;
        float v = c[r] + bv;
        if (RELU) v = fmaxf(v, 0.0f);
        if (OUT_MODE == 0)      Cf[(size_t)m * N + n] = v;
        else if (OUT_MODE == 1) Cb[packA_idx(m, n, N)] = (bf16_t)v;
        else                    Cb[(size_t)m * N + n] = (bf16_t)v;
    }
}

template<int OUT_MODE, bool HAS_BIAS, bool RELU>
__global__ void gemm_wmma_kernel(const bf16_t* __restrict__ Apk,
                                 const bf16_t* __restrict__ Bpk,
                                 const float* __restrict__ bias,
                                 float* __restrict__ Cf,
                                 bf16_t* __restrict__ Cb,
                                 int M, int N, int K) {
    // Force wave id into an SGPR: all tile indexing / guards become scalar.
    const int wave = __builtin_amdgcn_readfirstlane((int)threadIdx.x >> 5);
    const int lane = (int)threadIdx.x & 31;
    const int tiles_n = N >> 4;
    const int tiles_m = M >> 4;              // M is a multiple of 16
    const int ksteps  = K >> 5;

    const int bn = (int)blockIdx.x % tiles_n;
    const int bm = ((int)blockIdx.x / tiles_n) * 16 + wave * 4;  // first m-tile
    if (bm >= tiles_m) return;               // scalar branch

    const size_t kstride = 32 * 16;                      // elems per k-step
    const size_t astep   = (size_t)ksteps * kstride;     // elems per m-tile

    const bf16_t* bp = Bpk + ((size_t)(bn * ksteps) * 32 + lane) * 16;
    const bf16_t* ap = Apk + ((size_t)(bm * ksteps) * 32 + lane) * 16;

    v8f c0 = {}, c1 = {}, c2 = {}, c3 = {};

    // -------- software-pipelined k-loop (double-buffered fragments) -------
    v16bf bcur = *(const v16bf*)bp;
    v16bf a0   = *(const v16bf*)(ap);
    v16bf a1   = *(const v16bf*)(ap + astep);
    v16bf a2   = *(const v16bf*)(ap + 2 * astep);
    v16bf a3   = *(const v16bf*)(ap + 3 * astep);

#pragma unroll 2
    for (int s = 0; s < ksteps - 1; ++s) {
        bp += kstride;
        ap += kstride;
        v16bf bnxt = *(const v16bf*)bp;
        v16bf n0   = *(const v16bf*)(ap);
        v16bf n1   = *(const v16bf*)(ap + astep);
        v16bf n2   = *(const v16bf*)(ap + 2 * astep);
        v16bf n3   = *(const v16bf*)(ap + 3 * astep);

        c0 = __builtin_amdgcn_wmma_f32_16x16x32_bf16(false, a0, false, bcur,
                                                     (short)0, c0, false, false);
        c1 = __builtin_amdgcn_wmma_f32_16x16x32_bf16(false, a1, false, bcur,
                                                     (short)0, c1, false, false);
        c2 = __builtin_amdgcn_wmma_f32_16x16x32_bf16(false, a2, false, bcur,
                                                     (short)0, c2, false, false);
        c3 = __builtin_amdgcn_wmma_f32_16x16x32_bf16(false, a3, false, bcur,
                                                     (short)0, c3, false, false);
        bcur = bnxt; a0 = n0; a1 = n1; a2 = n2; a3 = n3;
    }
    c0 = __builtin_amdgcn_wmma_f32_16x16x32_bf16(false, a0, false, bcur,
                                                 (short)0, c0, false, false);
    c1 = __builtin_amdgcn_wmma_f32_16x16x32_bf16(false, a1, false, bcur,
                                                 (short)0, c1, false, false);
    c2 = __builtin_amdgcn_wmma_f32_16x16x32_bf16(false, a2, false, bcur,
                                                 (short)0, c2, false, false);
    c3 = __builtin_amdgcn_wmma_f32_16x16x32_bf16(false, a3, false, bcur,
                                                 (short)0, c3, false, false);

    const int half = lane >> 4;
    const int n = (bn << 4) + (lane & 15);
    store_subtile<OUT_MODE, HAS_BIAS, RELU>(c0, (bm + 0) << 4, n, half, bias, Cf, Cb, N);
    if ((bm + 1) < tiles_m)
        store_subtile<OUT_MODE, HAS_BIAS, RELU>(c1, (bm + 1) << 4, n, half, bias, Cf, Cb, N);
    if ((bm + 2) < tiles_m)
        store_subtile<OUT_MODE, HAS_BIAS, RELU>(c2, (bm + 2) << 4, n, half, bias, Cf, Cb, N);
    if ((bm + 3) < tiles_m)
        store_subtile<OUT_MODE, HAS_BIAS, RELU>(c3, (bm + 3) << 4, n, half, bias, Cf, Cb, N);
}

// ---------------------------------------------------------------------------
// Host-side orchestration
// ---------------------------------------------------------------------------
extern "C" void kernel_launch(void* const* d_in, const int* in_sizes, int n_in,
                              void* d_out, int out_size, void* d_ws, size_t ws_size,
                              hipStream_t stream) {
    (void)n_in; (void)out_size; (void)ws_size;

    const float*      x    = (const float*)d_in[0];
    const long long*  eidx = (const long long*)d_in[1];   // int64 [2,E]
    const float*      Wg   = (const float*)d_in[2];
    const float*      bg   = (const float*)d_in[3];
    const float*      W1   = (const float*)d_in[4];
    const float*      b1   = (const float*)d_in[5];
    const float*      W2   = (const float*)d_in[6];
    const float*      b2   = (const float*)d_in[7];
    const float*      W3   = (const float*)d_in[8];
    const float*      b3   = (const float*)d_in[9];
    float*            y    = (float*)d_out;

    const int N = in_sizes[0] / DGCN;
    const int E = in_sizes[1] / 2;
    const int H1 = 512, H2 = 256;

    const int tiles_m = N / 16;                     // N divisible by 16
    const int mblocks = (tiles_m + 15) / 16;        // blocks of 4 waves x 4 tiles
    const int padM    = mblocks * 16 * 16;          // rows incl. grid padding

    char* ws = (char*)d_ws;
    size_t off = 0;
    auto carve = [&](size_t bytes) -> char* {
        char* p = ws + off;
        off += (bytes + 255) & ~(size_t)255;
        return p;
    };
    // Packed-A buffers padded to the grid footprint (branch-free k-loop).
    bf16_t* xpk  = (bf16_t*)carve((size_t)padM * DGCN * 2);  // packed x / packed 'a'
    float*  h    = (float*) carve((size_t)N * DGCN * 4);     // x@Wg f32; reused as h2
    float*  outg = (float*) carve((size_t)N * DGCN * 4);     // GCN aggregate
    bf16_t* h1pk = (bf16_t*)carve((size_t)padM * H1 * 2);    // packed-A h1
    float*  deg  = (float*) carve((size_t)N * 4);
    bf16_t* Wgpk = (bf16_t*)carve((size_t)DGCN * DGCN * 2);
    bf16_t* W1pk = (bf16_t*)carve((size_t)DGCN * H1 * 2);
    bf16_t* W2pk = (bf16_t*)carve((size_t)H1 * H2 * 2);
    bf16_t* apk  = xpk;
    bf16_t* h2   = (bf16_t*)h;

    const long long* srcI = eidx;
    const long long* dstI = eidx + E;
    const size_t ND = (size_t)N * DGCN;

    // 1) degrees (self loop = 1) -> dinv (in place)
    set_ones_kernel<<<(N + 255) / 256, 256, 0, stream>>>(deg, N);
    deg_count_kernel<<<(E + 255) / 256, 256, 0, stream>>>(dstI, deg, E);
    rsqrt_inplace_kernel<<<(N + 255) / 256, 256, 0, stream>>>(deg, N);

    // 2) pack operands into WMMA fragment layouts (bf16)
    pack_a_f32_kernel<<<(unsigned)((ND + 255) / 256), 256, 0, stream>>>(x, xpk, N, DGCN);
    pack_b_f32_kernel<<<(DGCN * DGCN + 255) / 256, 256, 0, stream>>>(Wg, Wgpk, DGCN, DGCN);
    pack_b_f32_kernel<<<(DGCN * H1 + 255) / 256, 256, 0, stream>>>(W1, W1pk, DGCN, H1);
    pack_b_f32_kernel<<<(H1 * H2 + 255) / 256, 256, 0, stream>>>(W2, W2pk, H1, H2);

    // 3) h = x @ W_gcn   (f32 out, no bias/relu)
    gemm_wmma_kernel<0, false, false><<<(DGCN / 16) * mblocks, 128, 0, stream>>>(
        xpk, Wgpk, nullptr, h, nullptr, N, DGCN, DGCN);

    // 4) GCN aggregate: self-loop seed + edge scatter
    init_out_kernel<<<(unsigned)((ND + 255) / 256), 256, 0, stream>>>(h, deg, outg, ND);
    scatter_kernel<<<(E + 7) / 8, 256, 0, stream>>>(srcI, dstI, h, deg, outg, E);

    // 5) a = relu(outg + b_gcn) + x  -> packed-A bf16
    relu_residual_pack_kernel<<<(unsigned)((ND + 255) / 256), 256, 0, stream>>>(
        outg, bg, x, apk, ND);

    // 6) h1 = relu(a @ W1 + b1) -> packed-A bf16 (next K = 512)
    gemm_wmma_kernel<1, true, true><<<(H1 / 16) * mblocks, 128, 0, stream>>>(
        apk, W1pk, b1, nullptr, h1pk, N, H1, DGCN);

    // 7) h2 = relu(h1 @ W2 + b2) -> bf16 row-major
    gemm_wmma_kernel<2, true, true><<<(H2 / 16) * mblocks, 128, 0, stream>>>(
        h1pk, W2pk, b2, nullptr, h2, N, H2, H1);

    // 8) y = h2 @ W3 + b3
    final_dot_kernel<<<(N + 7) / 8, 256, 0, stream>>>(h2, W3, b3, y, N);
}
